// CAMSAAttention_56453050138964
// MI455X (gfx1250) — matrix operand order repaired and checked by reading, not compiled, not run on hardware
//
#include <hip/hip_runtime.h>

// ---------------- types ----------------
typedef __attribute__((ext_vector_type(16))) __bf16 v16bf;
typedef __attribute__((ext_vector_type(8)))  __bf16 v8bf;
typedef __attribute__((ext_vector_type(8)))  float  v8f;
typedef __attribute__((ext_vector_type(4)))  float  f4;
typedef __attribute__((ext_vector_type(4)))  int    i4;

// explicit global-address-space pointers so GEMM loads lower to global_load_*
// (generic pointers through loop PHIs otherwise fall back to flat_load_*,
//  which ticks DScnt as well as LOADcnt per the CDNA5 ISA)
typedef const __bf16 __attribute__((address_space(1))) * gbfp;
typedef const v8bf   __attribute__((address_space(1))) * gv8p;

union Frag16 { v16bf f; v8bf h[2]; };

static __device__ __forceinline__ __bf16 f2bf(float x) {
    unsigned u = __builtin_bit_cast(unsigned, x);
    unsigned r = (u + 0x7FFFu + ((u >> 16) & 1u)) >> 16;   // round-to-nearest-even
    unsigned short hs = (unsigned short)r;
    return __builtin_bit_cast(__bf16, hs);
}

// Problem constants
constexpr int Bb = 4, Ss = 2048, Dd = 1024, Mm = 4;
constexpr size_t MBv = 1u << 20;

// ---------------- elementwise f32 -> bf16 ----------------
__global__ void f32_to_bf16_kernel(const float* __restrict__ in,
                                   __bf16* __restrict__ out, size_t n) {
    size_t i = (size_t)blockIdx.x * blockDim.x + threadIdx.x;
    if (i < n) out[i] = f2bf(in[i]);
}

// ---------------- convert + transpose (W [D,D] f32 -> Wt [D,D] bf16) ----------------
__global__ void convert_transpose_kernel(const float* __restrict__ in,
                                         __bf16* __restrict__ out,
                                         int rows, int cols) {
    __shared__ float tile[32][33];
    int tx = threadIdx.x, ty = threadIdx.y;
    int cBase = blockIdx.x * 32, rBase = blockIdx.y * 32;
    #pragma unroll
    for (int j = 0; j < 4; ++j)
        tile[ty + j * 8][tx] = in[(size_t)(rBase + ty + j * 8) * cols + cBase + tx];
    __syncthreads();
    #pragma unroll
    for (int j = 0; j < 4; ++j)
        out[(size_t)(cBase + ty + j * 8) * rows + rBase + tx] = f2bf(tile[tx][ty + j * 8]);
}

// ---------------- bf16 transpose (batched): in [Z,rows,cols] -> out [Z,cols,rows] ----
__global__ void transpose_bf16_kernel(const __bf16* __restrict__ in,
                                      __bf16* __restrict__ out,
                                      int rows, int cols) {
    __shared__ __bf16 tile[32][33];
    const __bf16* inb = in + (size_t)blockIdx.z * rows * cols;
    __bf16* outb      = out + (size_t)blockIdx.z * rows * cols;
    int tx = threadIdx.x, ty = threadIdx.y;
    int cBase = blockIdx.x * 32, rBase = blockIdx.y * 32;
    #pragma unroll
    for (int j = 0; j < 4; ++j)
        tile[ty + j * 8][tx] = inb[(size_t)(rBase + ty + j * 8) * cols + cBase + tx];
    __syncthreads();
    #pragma unroll
    for (int j = 0; j < 4; ++j)
        outb[(size_t)(cBase + ty + j * 8) * rows + rBase + tx] = tile[tx][ty + j * 8];
}

// ---------------- WMMA NT GEMM: C[M,N] = alpha * A[M,K] * Bt[N,K]^T (+ bias[N]) ------
// Block = 256 threads = 8 waves arranged 2(M) x 4(N); block tile 128 x 128.
// Wave tile 64(M) x 32(N) = 4x2 WMMA tiles: 8 WMMAs per 6 fragment loads.
// Six loop-carried per-lane AS(1) row pointers; K advance folds into immediate offsets.
// Fragment layouts follow CDNA5 ISA 7.12.2 (16-bit A 16x32 / B 32x16, wave32).
template <bool OUT_BF16, bool HAS_BIAS>
__global__ void __launch_bounds__(256, 1)
gemm_nt_kernel(const __bf16* __restrict__ A, size_t strideA,
               const __bf16* __restrict__ Bt, size_t strideB,
               void* __restrict__ C, size_t strideC,
               const float* __restrict__ bias,
               int Kdim, int ldc, float alpha) {
    const int lane = threadIdx.x & 31;
    const int wave = threadIdx.x >> 5;
    const int hf   = lane >> 4;      // half-wave select
    const int r    = lane & 15;

    const int waveM = wave & 1;      // 2 waves along M (64 rows each)
    const int waveN = wave >> 1;     // 4 waves along N (32 cols each)
    const int m0 = blockIdx.y * 128 + waveM * 64;
    const int n0 = blockIdx.x * 128 + waveN * 32;

    const __bf16* __restrict__ Ab  = A  + (size_t)blockIdx.z * strideA;
    const __bf16* __restrict__ Btb = Bt + (size_t)blockIdx.z * strideB;

    // loop-carried per-lane row pointers in the GLOBAL address space
    gbfp ap[4];
    gbfp bp[2];
    #pragma unroll
    for (int mi = 0; mi < 4; ++mi) ap[mi] = (gbfp)(Ab  + (size_t)(m0 + mi * 16 + r) * Kdim + hf * 8);
    #pragma unroll
    for (int ni = 0; ni < 2; ++ni) bp[ni] = (gbfp)(Btb + (size_t)(n0 + ni * 16 + r) * Kdim + hf * 8);

    v8f acc[4][2] = {};

    for (int k = 0; k < Kdim; k += 32) {
        Frag16 a[4], b[2];
        #pragma unroll
        for (int mi = 0; mi < 4; ++mi) {
            a[mi].h[0] = *(gv8p)(ap[mi]);        // K 0-7  / 8-15
            a[mi].h[1] = *(gv8p)(ap[mi] + 16);   // K 16-23 / 24-31
            ap[mi] += 32;
        }
        #pragma unroll
        for (int ni = 0; ni < 2; ++ni) {
            b[ni].h[0] = *(gv8p)(bp[ni]);        // K 0-7  / 8-15
            b[ni].h[1] = *(gv8p)(bp[ni] + 16);   // K 16-23 / 24-31
            bp[ni] += 32;
        }
        #pragma unroll
        for (int mi = 0; mi < 4; ++mi)
            #pragma unroll
            for (int ni = 0; ni < 2; ++ni)
                acc[mi][ni] = __builtin_amdgcn_wmma_f32_16x16x32_bf16(
                    false, a[mi].f, false, b[ni].f, (short)0, acc[mi][ni], false, false);
    }

    // writeback: lane holds column n, rows (hf*8 .. hf*8+7) within each 16x16 tile
    #pragma unroll
    for (int mi = 0; mi < 4; ++mi) {
        #pragma unroll
        for (int ni = 0; ni < 2; ++ni) {
            const int n = n0 + ni * 16 + r;
            float bval = 0.0f;
            if (HAS_BIAS) bval = bias[n];
            #pragma unroll
            for (int i = 0; i < 8; ++i) {
                const int m = m0 + mi * 16 + hf * 8 + i;
                const float v = acc[mi][ni][i] * alpha + bval;
                const size_t idx = (size_t)blockIdx.z * strideC + (size_t)m * ldc + n;
                if (OUT_BF16) reinterpret_cast<__bf16*>(C)[idx] = f2bf(v);
                else          reinterpret_cast<float*>(C)[idx]  = v;
            }
        }
    }
}

// ---------------- fused 4-mask softmax-sum: P[b,q,:] = sum_m softmax(mask_m) -------
// One 256-thread block per (q,b) row; thread t owns columns 8t..8t+7 (vectorized).
__global__ void mask_softmax_sum_kernel(const float* __restrict__ scores,
                                        const int* __restrict__ masks,
                                        __bf16* __restrict__ P) {
    const int q = blockIdx.x;     // 0..S-1
    const int b = blockIdx.y;     // batch
    const int t = threadIdx.x;    // 0..255
    const f4* srow = reinterpret_cast<const f4*>(scores + ((size_t)b * Ss + q) * Ss);

    float s[8], pacc[8];
    {
        f4 s0 = srow[t * 2], s1 = srow[t * 2 + 1];
        #pragma unroll
        for (int j = 0; j < 4; ++j) { s[j] = s0[j]; s[4 + j] = s1[j]; pacc[j] = 0.0f; pacc[4 + j] = 0.0f; }
    }

    __shared__ float red[256];

    for (int m = 0; m < Mm; ++m) {
        const i4* mrow = reinterpret_cast<const i4*>(masks + ((size_t)m * Ss + q) * Ss);
        i4 m0 = mrow[t * 2], m1 = mrow[t * 2 + 1];
        float mv[8], lmax = -3.0e38f;
        #pragma unroll
        for (int j = 0; j < 4; ++j) {
            mv[j]     = m0[j] ? s[j]     : -1.0e9f;
            mv[4 + j] = m1[j] ? s[4 + j] : -1.0e9f;
        }
        #pragma unroll
        for (int j = 0; j < 8; ++j) lmax = fmaxf(lmax, mv[j]);

        red[t] = lmax; __syncthreads();
        for (int off = 128; off > 0; off >>= 1) {
            if (t < off) red[t] = fmaxf(red[t], red[t + off]);
            __syncthreads();
        }
        const float rmax = red[0]; __syncthreads();

        float e[8], lsum = 0.0f;
        #pragma unroll
        for (int j = 0; j < 8; ++j) { e[j] = __expf(mv[j] - rmax); lsum += e[j]; }
        red[t] = lsum; __syncthreads();
        for (int off = 128; off > 0; off >>= 1) {
            if (t < off) red[t] += red[t + off];
            __syncthreads();
        }
        const float inv = 1.0f / red[0]; __syncthreads();

        #pragma unroll
        for (int j = 0; j < 8; ++j) pacc[j] += e[j] * inv;
    }

    union { v8bf v; __bf16 e[8]; } pv;
    #pragma unroll
    for (int j = 0; j < 8; ++j) pv.e[j] = f2bf(pacc[j]);
    reinterpret_cast<v8bf*>(P + ((size_t)b * Ss + q) * Ss)[t] = pv.v;
}

// ---------------- launch ----------------
extern "C" void kernel_launch(void* const* d_in, const int* in_sizes, int n_in,
                              void* d_out, int out_size, void* d_ws, size_t ws_size,
                              hipStream_t stream) {
    const float* x     = (const float*)d_in[0];
    const int*   masks = (const int*)d_in[1];
    const float* Wq = (const float*)d_in[2]; const float* bq = (const float*)d_in[3];
    const float* Wk = (const float*)d_in[4]; const float* bk = (const float*)d_in[5];
    const float* Wv = (const float*)d_in[6]; const float* bv = (const float*)d_in[7];
    const float* Wo = (const float*)d_in[8]; const float* bo = (const float*)d_in[9];
    float* out = (float*)d_out;

    char* ws = (char*)d_ws;
    __bf16* xb   = (__bf16*)(ws + 0);           // 16 MB  [B*S, D]
    __bf16* WqT  = (__bf16*)(ws + 16 * MBv);    //  2 MB  [D, D] transposed
    __bf16* WkT  = (__bf16*)(ws + 18 * MBv);
    __bf16* WvT  = (__bf16*)(ws + 20 * MBv);
    __bf16* WoT  = (__bf16*)(ws + 22 * MBv);
    __bf16* Qb   = (__bf16*)(ws + 24 * MBv);    // 16 MB  [B,S,D]
    __bf16* Kb   = (__bf16*)(ws + 40 * MBv);    // 16 MB
    __bf16* Vb   = (__bf16*)(ws + 56 * MBv);    // 16 MB
    __bf16* VTb  = (__bf16*)(ws + 72 * MBv);    // 16 MB  [B,D,S]
    __bf16* attn = (__bf16*)(ws + 88 * MBv);    // 16 MB  [B,S,D]
    float*  sc   = (float*) (ws + 104 * MBv);   // 64 MB  [B,S,S]
    __bf16* Pb   = (__bf16*)(ws + 168 * MBv);   // 32 MB  [B,S,S]

    const size_t nX = (size_t)Bb * Ss * Dd;     // 8.4M
    const float invSqrtD = 0.03125f;            // 1/sqrt(1024)
    const float invM = 0.25f;

    // 1) x -> bf16 ; W* -> bf16 transposed
    f32_to_bf16_kernel<<<dim3((unsigned)((nX + 255) / 256)), 256, 0, stream>>>(x, xb, nX);
    dim3 tb(32, 8), tg(Dd / 32, Dd / 32);
    convert_transpose_kernel<<<tg, tb, 0, stream>>>(Wq, WqT, Dd, Dd);
    convert_transpose_kernel<<<tg, tb, 0, stream>>>(Wk, WkT, Dd, Dd);
    convert_transpose_kernel<<<tg, tb, 0, stream>>>(Wv, WvT, Dd, Dd);
    convert_transpose_kernel<<<tg, tb, 0, stream>>>(Wo, WoT, Dd, Dd);

    // 2) Q/K/V = x*W + b   (M=B*S=8192, N=1024, K=1024)
    dim3 gQKV(Dd / 128, (Bb * Ss) / 128, 1);
    gemm_nt_kernel<true, true><<<gQKV, 256, 0, stream>>>(xb, 0, WqT, 0, Qb, 0, bq, Dd, Dd, 1.0f);
    gemm_nt_kernel<true, true><<<gQKV, 256, 0, stream>>>(xb, 0, WkT, 0, Kb, 0, bk, Dd, Dd, 1.0f);
    gemm_nt_kernel<true, true><<<gQKV, 256, 0, stream>>>(xb, 0, WvT, 0, Vb, 0, bv, Dd, Dd, 1.0f);

    // 3) V^T per batch  [S,D] -> [D,S]
    transpose_bf16_kernel<<<dim3(Dd / 32, Ss / 32, Bb), tb, 0, stream>>>(Vb, VTb, Ss, Dd);

    // 4) scores = Q K^T / sqrt(D)   (batched, M=N=2048, K=1024, f32 out)
    dim3 gS(Ss / 128, Ss / 128, Bb);
    gemm_nt_kernel<false, false><<<gS, 256, 0, stream>>>(
        Qb, (size_t)Ss * Dd, Kb, (size_t)Ss * Dd, sc, (size_t)Ss * Ss, nullptr, Dd, Ss, invSqrtD);

    // 5) P = sum_m softmax(mask_m(scores))   (one block per (q,b) row)
    mask_softmax_sum_kernel<<<dim3(Ss, Bb), 256, 0, stream>>>(sc, masks, Pb);

    // 6) attn = (P @ V) / M   (batched, M=2048, N=1024, K=2048, bf16 out)
    dim3 gA(Dd / 128, Ss / 128, Bb);
    gemm_nt_kernel<true, false><<<gA, 256, 0, stream>>>(
        Pb, (size_t)Ss * Ss, VTb, (size_t)Dd * Ss, attn, (size_t)Ss * Dd, nullptr, Ss, Dd, invM);

    // 7) out = attn @ Wo + bo   (M=8192, N=1024, K=1024, f32 out)
    gemm_nt_kernel<false, true><<<gQKV, 256, 0, stream>>>(
        attn, 0, WoT, 0, out, 0, bo, Dd, Dd, 1.0f);
}
